// RBFKernel_65481071402842
// MI455X (gfx1250) — matrix operand order repaired and checked by reading, not compiled, not run on hardware
//
#include <hip/hip_runtime.h>
#include <hip/hip_bf16.h>

// ---------------------------------------------------------------------------
// RBF kernel for MI455X (gfx1250, wave32, WMMA).
//
// cross = x @ y^T computed as a bf16 WMMA GEMM with hi/lo fp32 splitting:
//   A' = [x_hi | x_hi | x_lo]  (8192 x 768 bf16)
//   B' = [y_hi | y_lo | y_hi]  (8192 x 768 bf16)
//   A' @ B'^T = x_hi@y_hi^T + x_hi@y_lo^T + x_lo@y_hi^T  ~= fp32 x@y^T
// Epilogue: out = exp(-gamma * max(xsq + ysq - 2*cross, 0)), NT fp32 stores.
//
// Global->LDS staging uses the CDNA5 async copy engine
// (GLOBAL_LOAD_ASYNC_TO_LDS_B128, ASYNCcnt) when the builtin exists;
// otherwise falls back to VGPR staging + ds_store_b128.
// ---------------------------------------------------------------------------

typedef __bf16 bf16;
typedef __attribute__((ext_vector_type(16))) __bf16 v16bf;
typedef __attribute__((ext_vector_type(8)))  __bf16 bf16x8;
typedef __attribute__((ext_vector_type(8)))  float  v8f;
typedef __attribute__((ext_vector_type(4)))  int    v4i;

#define NROWS 8192      // N == M
#define DDIM  256
#define KP    768       // packed K' = 3*D (bf16)
#define BLK   128       // C tile is BLK x BLK
#define KT    64        // K-tile (bf16 elements) per LDS stage
#define KSTEPS (KP / KT) // 12
#define LDK   72        // padded LDS row pitch (elems): 144B -> conflict-free

#if defined(__HIP_DEVICE_COMPILE__) && __has_builtin(__builtin_amdgcn_global_load_async_to_lds_b128)
#define HAVE_ASYNC_COPY 1
#else
#define HAVE_ASYNC_COPY 0
#endif

union V16 { v16bf v; bf16x8 h[2]; };

#if HAVE_ASYNC_COPY
// Builtin signature (from hipcc diagnostic): first param is int4* (global),
// second is int4* (LDS), then imm offset + imm cpol.
typedef __attribute__((address_space(1))) v4i gv4i;
typedef __attribute__((address_space(3))) v4i lv4i;

// 16B global -> LDS copy through the async DMA path (no VGPR staging).
// Generic LDS pointers carry the segment offset in addr[31:0] (flat-aperture
// rule), so an integer truncation yields the addrspace(3) address.
__device__ __forceinline__ void async_cp16(const bf16* gsrc, bf16* ldst) {
    __builtin_amdgcn_global_load_async_to_lds_b128(
        (gv4i*)(unsigned long long)gsrc,
        (lv4i*)(unsigned int)(unsigned long long)ldst,
        0, 0);
}
__device__ __forceinline__ void wait_async0() {
#if __has_builtin(__builtin_amdgcn_s_wait_asynccnt)
    __builtin_amdgcn_s_wait_asynccnt(0);
#else
    asm volatile("s_wait_asynccnt 0" ::: "memory");
#endif
}
#endif

// --------------------------- prep: split + norms ---------------------------
// One wave per row. mode 0: [hi|hi|lo] (for x). mode 1: [hi|lo|hi] (for y).
__global__ __launch_bounds__(256)
void rbf_prep_kernel(const float* __restrict__ src, bf16* __restrict__ dst,
                     float* __restrict__ sq, int mode)
{
    const int lane = threadIdx.x & 31;
    const int wave = threadIdx.x >> 5;
    const int row  = blockIdx.x * 8 + wave;

    const float* s = src + (size_t)row * DDIM;
    bf16*        d = dst + (size_t)row * KP;

    float acc = 0.0f;
    for (int c = lane; c < DDIM; c += 32) {
        float f = s[c];
        bf16  h = (bf16)f;            // RNE truncation to bf16
        float r = f - (float)h;       // exact residual
        bf16  l = (bf16)r;
        d[c] = h;
        if (mode == 0) { d[DDIM + c] = h; d[2 * DDIM + c] = l; }
        else           { d[DDIM + c] = l; d[2 * DDIM + c] = h; }
        acc += f * f;
    }
    // wave32 reduction
    for (int off = 16; off > 0; off >>= 1)
        acc += __shfl_xor(acc, off, 32);
    if (lane == 0) sq[row] = acc;
}

// ------------------------------- main GEMM ---------------------------------
__global__ __launch_bounds__(256)
void rbf_gemm_kernel(const bf16* __restrict__ Xp, const bf16* __restrict__ Yp,
                     const float* __restrict__ xsq, const float* __restrict__ ysq,
                     const float* __restrict__ gptr, float* __restrict__ out)
{
    // double-buffered A/B tiles: 2 bufs * 2 mats * 128 rows * 72 elems * 2B = 72KB
    __shared__ bf16 smem[2 * 2 * BLK * LDK];

    const int tid  = threadIdx.x;
    const int lane = tid & 31;
    const int wave = tid >> 5;      // 8 waves
    const int wm   = wave >> 1;     // 0..3 : 32-row band
    const int wn   = wave & 1;      // 0..1 : 64-col band

    const int m0 = blockIdx.y * BLK;
    const int n0 = blockIdx.x * BLK;

    // global->LDS staging: 2 threads per tile row, 32 cols (64B) each
    const int lrow = tid >> 1;           // 0..127
    const int lcol = (tid & 1) * 32;     // 0 or 32

    const bf16* gA = Xp + (size_t)(m0 + lrow) * KP + lcol;
    const bf16* gB = Yp + (size_t)(n0 + lrow) * KP + lcol;

    v8f acc[2][4];
    const v8f vzero = {0.f, 0.f, 0.f, 0.f, 0.f, 0.f, 0.f, 0.f};
#pragma unroll
    for (int i = 0; i < 2; ++i)
#pragma unroll
        for (int j = 0; j < 4; ++j) acc[i][j] = vzero;

    auto compute = [&](int buf) {
        const bf16* sA = smem + buf * (2 * BLK * LDK);
        const bf16* sB = sA + BLK * LDK;
        const int lm = lane & 15;
        // ISA 16-bit A 16x32 layout: lanes 0-15 hold K {0..7,16..23}, lanes
        // 16-31 hold K {8..15,24..31}; B 32x16: same K order as A
        const int ah = (lane & 16) ? 8 : 0;
        const int bh = (lane & 16) ? 8 : 0;
#pragma unroll
        for (int s = 0; s < 2; ++s) {          // two K=32 sub-steps per K-tile
            V16 a[2], b[4];
#pragma unroll
            for (int mt = 0; mt < 2; ++mt) {
                const bf16* p = sA + (wm * 32 + mt * 16 + lm) * LDK + s * 32 + ah;
                a[mt].h[0] = *(const bf16x8*)(p);
                a[mt].h[1] = *(const bf16x8*)(p + 16);
            }
#pragma unroll
            for (int nt = 0; nt < 4; ++nt) {
                const bf16* p = sB + (wn * 64 + nt * 16 + lm) * LDK + s * 32 + bh;
                b[nt].h[0] = *(const bf16x8*)(p);
                b[nt].h[1] = *(const bf16x8*)(p + 16);
            }
#pragma unroll
            for (int mt = 0; mt < 2; ++mt)
#pragma unroll
                for (int nt = 0; nt < 4; ++nt)
                    acc[mt][nt] = __builtin_amdgcn_wmma_f32_16x16x32_bf16(
                        false, a[mt].v, false, b[nt].v,
                        (short)0, acc[mt][nt], false, false);
        }
    };

#if HAVE_ASYNC_COPY
    // -------- async DMA staging: no VGPR round trip, tracked by ASYNCcnt ----
    auto stage_async = [&](int buf, int kbase) {
        bf16* sA = smem + buf * (2 * BLK * LDK);
        bf16* sB = sA + BLK * LDK;
#pragma unroll
        for (int j = 0; j < 4; ++j) {
            async_cp16(gA + kbase + j * 8, sA + lrow * LDK + lcol + j * 8);
            async_cp16(gB + kbase + j * 8, sB + lrow * LDK + lcol + j * 8);
        }
    };

    stage_async(0, 0);
    wait_async0();
    __syncthreads();

    int buf = 0;
    for (int kt = 0; kt < KSTEPS; ++kt) {
        if (kt + 1 < KSTEPS) stage_async(buf ^ 1, (kt + 1) * KT);  // overlap DMA
        compute(buf);
        if (kt + 1 < KSTEPS) {
            wait_async0();      // own wave's DMA done
            __syncthreads();    // everyone's DMA visible
            buf ^= 1;
        }
    }
#else
    // -------- fallback: global->VGPR->ds_store staging ---------------------
    bf16x8 ra[4], rb[4];
    auto load_regs = [&](int kbase) {
#pragma unroll
        for (int j = 0; j < 4; ++j) {
            ra[j] = *(const bf16x8*)(gA + kbase + j * 8);
            rb[j] = *(const bf16x8*)(gB + kbase + j * 8);
        }
    };
    auto store_lds = [&](int buf) {
        bf16* sA = smem + buf * (2 * BLK * LDK);
        bf16* sB = sA + BLK * LDK;
#pragma unroll
        for (int j = 0; j < 4; ++j) {
            *(bf16x8*)(sA + lrow * LDK + lcol + j * 8) = ra[j];
            *(bf16x8*)(sB + lrow * LDK + lcol + j * 8) = rb[j];
        }
    };

    load_regs(0);
    store_lds(0);
    __syncthreads();

    int buf = 0;
    for (int kt = 0; kt < KSTEPS; ++kt) {
        if (kt + 1 < KSTEPS) load_regs((kt + 1) * KT);
        compute(buf);
        if (kt + 1 < KSTEPS) {
            store_lds(buf ^ 1);
            __syncthreads();
            buf ^= 1;
        }
    }
#endif

    // ------------------------------ epilogue -------------------------------
    const float gamma = *gptr;
    const int lm = lane & 15;
    const int mh = (lane & 16) ? 8 : 0;   // C/D layout: lanes 16-31 hold M+8
#pragma unroll
    for (int mt = 0; mt < 2; ++mt) {
        const int gmBase = m0 + wm * 32 + mt * 16 + mh;
        float xs[8];
#pragma unroll
        for (int e = 0; e < 8; ++e) xs[e] = xsq[gmBase + e];
#pragma unroll
        for (int nt = 0; nt < 4; ++nt) {
            const int gn = n0 + wn * 64 + nt * 16 + lm;
            const float ys = ysq[gn];
            v8f c = acc[mt][nt];
#pragma unroll
            for (int e = 0; e < 8; ++e) {
                float d2 = xs[e] + ys - 2.0f * c[e];
                d2 = fmaxf(d2, 0.0f);
                float r = __expf(-gamma * d2);
                // output is written once, never re-read: keep it out of caches
                __builtin_nontemporal_store(
                    r, out + (size_t)(gmBase + e) * NROWS + gn);
            }
        }
    }
}

// ------------------------------- launcher ----------------------------------
extern "C" void kernel_launch(void* const* d_in, const int* in_sizes, int n_in,
                              void* d_out, int out_size, void* d_ws, size_t ws_size,
                              hipStream_t stream) {
    const float* x = (const float*)d_in[0];
    const float* y = (const float*)d_in[1];
    const float* g = (const float*)d_in[2];
    float* out = (float*)d_out;

    char* ws = (char*)d_ws;
    const size_t packed_bytes = (size_t)NROWS * KP * sizeof(bf16); // 12 MB each
    bf16*  Xp  = (bf16*)ws;
    bf16*  Yp  = (bf16*)(ws + packed_bytes);
    float* xsq = (float*)(ws + 2 * packed_bytes);
    float* ysq = xsq + NROWS;

    // split/pack + norms (one wave per row, 8 rows per block)
    rbf_prep_kernel<<<NROWS / 8, 256, 0, stream>>>(x, Xp, xsq, 0);
    rbf_prep_kernel<<<NROWS / 8, 256, 0, stream>>>(y, Yp, ysq, 1);

    dim3 grid(NROWS / BLK, NROWS / BLK);   // 64 x 64 blocks
    rbf_gemm_kernel<<<grid, 256, 0, stream>>>(Xp, Yp, xsq, ysq, g, out);
}
